// LEMCellODE2_73254962200836
// MI455X (gfx1250) — compile-verified, matrix operand order checked by reading, not run on hardware
//
#include <hip/hip_runtime.h>
#include <hip/hip_bf16.h>

// LEM cell fused forward for MI455X (gfx1250, wave32, WMMA f16->f32).
// Gates kernel runs two phased K-loops (x@W_ih 4 gates, then y@W_hh 3 gates)
// so every wmma accumulator is touched once per K-step (no RAW copy churn),
// with double-buffered LDS weight staging and software-pipelined loads.

#define BATCH 16384
#define NINP  1024
#define NHID  1024
#define KSTEP 32

typedef __attribute__((ext_vector_type(16))) _Float16 v16h;
typedef __attribute__((ext_vector_type(4)))  _Float16 v4h;
typedef __attribute__((ext_vector_type(8)))  float    v8f;
typedef __attribute__((ext_vector_type(4)))  unsigned int v4u;

union FragU { v4u u[2]; v16h h; };

static __device__ __forceinline__ v4u ldq(const _Float16* p) {
  return *(const v4u*)p;
}
static __device__ __forceinline__ v16h make_frag(v4u lo, v4u hi) {
  FragU f; f.u[0] = lo; f.u[1] = hi; return f.h;
}
static __device__ __forceinline__ v8f wmma_(v16h a, v16h b, v8f c) {
  return __builtin_amdgcn_wmma_f32_16x16x32_f16(false, a, false, b,
                                                (short)0, c, false, false);
}
static __device__ __forceinline__ float sigmoidf_(float x) {
  return 1.0f / (1.0f + __expf(-x));
}
static __device__ __forceinline__ float tanhf_(float x) {
  float a = fabsf(x);
  float e = __expf(-2.0f * a);
  float t = (1.0f - e) / (1.0f + e);
  return copysignf(t, x);
}

// ---------------- conversion: f32 -> f16, 4 elems/thread ----------------
__global__ void cvt_f32_f16_x4(const float* __restrict__ in,
                               _Float16* __restrict__ out, int n4) {
  int i = blockIdx.x * blockDim.x + threadIdx.x;
  if (i < n4) {
    float4 v = ((const float4*)in)[i];
    v4h h = { (_Float16)v.x, (_Float16)v.y, (_Float16)v.z, (_Float16)v.w };
    ((v4h*)out)[i] = h;
  }
}

// ---------------- dt gate scalars: sigmoid(dt*Wdt + bdt) ----------------
__global__ void dt_gate_kernel(const float* __restrict__ dt,
                               const float* __restrict__ Wdt,
                               const float* __restrict__ bdt,
                               float* __restrict__ sig1,
                               float* __restrict__ sig2) {
  int i = blockIdx.x * blockDim.x + threadIdx.x;
  if (i < BATCH) {
    float d = dt[i];
    sig1[i] = sigmoidf_(d * Wdt[0] + bdt[0]);
    sig2[i] = sigmoidf_(d * Wdt[1] + bdt[1]);
  }
}

// ---------------- fused gates: 7 GEMM contributions + z_new -------------
// grid (BATCH/256, NHID/16), block 256 = 8 waves.
// Wave w owns M rows [m0+32w, m0+32w+32) as two 16-row tiles; all waves
// share one 16-column N tile. Phase 1: x@{W1,W2,W3,W4}. Phase 2: y@{U1,U2,U3}.
__global__ __launch_bounds__(256) void lem_gates_kernel(
    const _Float16* __restrict__ xh, const _Float16* __restrict__ yh,
    const _Float16* __restrict__ Wih, const _Float16* __restrict__ Whh,
    const float* __restrict__ b_ih, const float* __restrict__ b_hh,
    const float* __restrict__ b_z, const float* __restrict__ z,
    const float* __restrict__ sig1, const float* __restrict__ sig2,
    float* __restrict__ z_new, _Float16* __restrict__ zh,
    _Float16* __restrict__ iz_st, _Float16* __restrict__ msb_st) {
  __shared__ __align__(16) _Float16 sB[2][4][16][KSTEP];  // 8 KB ping-pong
  v4u* sbq = (v4u*)&sB[0][0][0][0];  // 256 v4u chunks per buffer

  const int tid  = threadIdx.x;
  const int wave = tid >> 5;
  const int lane = tid & 31;
  const int ksel = lane >> 4;   // 0: K 0-7/16-23, 1: K 8-15/24-31
  const int lrow = lane & 15;

  const int n0 = blockIdx.y * 16;
  const int m0 = blockIdx.x * 256 + wave * 32;
  const int NIT = NINP / KSTEP;

  v8f a1_0 = {}; v8f a1_1 = {}; v8f a2_0 = {}; v8f a2_1 = {};
  v8f iz_0 = {}; v8f iz_1 = {}; v8f ayc0 = {}; v8f ayc1 = {};

  int buf = 0;

  // ================= Phase 1: x @ {W1,W2,W3,W4} =================
  {
    // chunk id == tid == v4u index within buffer (4*16*4 == 256 chunks)
    const int g = tid >> 6, r = (tid >> 2) & 15, q = tid & 3;
    const _Float16* p =
        Wih + ((size_t)g * NHID + n0 + r) * NINP + q * 8;

    const _Float16* xr0 = xh + (size_t)(m0 + lrow) * NINP + ksel * 8;
    const _Float16* xr1 = xr0 + 16 * NINP;

    sbq[tid] = ldq(p);
    __syncthreads();

    v16h ax0 = make_frag(ldq(xr0), ldq(xr0 + 16));
    v16h ax1 = make_frag(ldq(xr1), ldq(xr1 + 16));

    for (int it = 0; it < NIT; ++it) {
      const bool more = (it + 1) < NIT;
      v4u t;
      v16h nax0, nax1;
      if (more) {
        p += KSTEP; t = ldq(p);
        xr0 += KSTEP; xr1 += KSTEP;
        nax0 = make_frag(ldq(xr0), ldq(xr0 + 16));
        nax1 = make_frag(ldq(xr1), ldq(xr1 + 16));
      }

      const int bb = buf * 256 + lrow * 4 + ksel;
#define BFRAG(g_) make_frag(sbq[bb + (g_) * 64], sbq[bb + (g_) * 64 + 2])
      v16h b;
      b = BFRAG(0); a1_0 = wmma_(ax0, b, a1_0); a1_1 = wmma_(ax1, b, a1_1);
      b = BFRAG(1); a2_0 = wmma_(ax0, b, a2_0); a2_1 = wmma_(ax1, b, a2_1);
      b = BFRAG(2); iz_0 = wmma_(ax0, b, iz_0); iz_1 = wmma_(ax1, b, iz_1);
      b = BFRAG(3); ayc0 = wmma_(ax0, b, ayc0); ayc1 = wmma_(ax1, b, ayc1);

      if (more) {
        sbq[(buf ^ 1) * 256 + tid] = t;
        ax0 = nax0; ax1 = nax1;
      }
      __syncthreads();
      buf ^= 1;
    }
  }

  // ================= Phase 2: y @ {U1,U2,U3} =================
  {
    const bool hv = tid < 192;           // 3*16*4 == 192 chunks
    const int c = hv ? tid : 0;
    const int g = c >> 6, r = (c >> 2) & 15, q = c & 3;
    const _Float16* p =
        Whh + ((size_t)g * NHID + n0 + r) * NHID + q * 8;

    const _Float16* yr0 = yh + (size_t)(m0 + lrow) * NHID + ksel * 8;
    const _Float16* yr1 = yr0 + 16 * NHID;

    if (hv) sbq[buf * 256 + tid] = ldq(p);
    __syncthreads();

    v16h ay0 = make_frag(ldq(yr0), ldq(yr0 + 16));
    v16h ay1 = make_frag(ldq(yr1), ldq(yr1 + 16));

    for (int it = 0; it < NIT; ++it) {
      const bool more = (it + 1) < NIT;
      v4u t;
      v16h nay0, nay1;
      if (more) {
        p += KSTEP;
        t = ldq(p);
        yr0 += KSTEP; yr1 += KSTEP;
        nay0 = make_frag(ldq(yr0), ldq(yr0 + 16));
        nay1 = make_frag(ldq(yr1), ldq(yr1 + 16));
      }

      const int bb = buf * 256 + lrow * 4 + ksel;
      v16h b;
      b = BFRAG(0); a1_0 = wmma_(ay0, b, a1_0); a1_1 = wmma_(ay1, b, a1_1);
      b = BFRAG(1); a2_0 = wmma_(ay0, b, a2_0); a2_1 = wmma_(ay1, b, a2_1);
      b = BFRAG(2); ayc0 = wmma_(ay0, b, ayc0); ayc1 = wmma_(ay1, b, ayc1);
#undef BFRAG

      if (more) {
        if (hv) sbq[(buf ^ 1) * 256 + tid] = t;
        ay0 = nay0; ay1 = nay1;
      }
      __syncthreads();
      buf ^= 1;
    }
  }

  // epilogue: C layout -> element (mtile_m0 + r + 8*ksel, n0 + lrow)
  const int nl = n0 + lrow;
  const float bi1 = b_ih[nl] + b_hh[nl];
  const float bi2 = b_ih[NHID + nl] + b_hh[NHID + nl];
  const float biy = b_ih[3 * NHID + nl] + b_hh[2 * NHID + nl];
  const float biz = b_ih[2 * NHID + nl] + b_z[nl];

#pragma unroll
  for (int r = 0; r < 8; ++r) {
    {
      const int m = m0 + r + 8 * ksel;
      const size_t o = (size_t)m * NHID + nl;
      float msb = sig1[m] * sigmoidf_(a1_0[r] + bi1);
      float ms  = sig2[m] * sigmoidf_(a2_0[r] + bi2);
      float t   = tanhf_(ayc0[r] + biy);
      float zn  = (1.0f - ms) * z[o] + ms * t;
      z_new[o]  = zn;
      zh[o]     = (_Float16)zn;
      iz_st[o]  = (_Float16)(iz_0[r] + biz);
      msb_st[o] = (_Float16)msb;
    }
    {
      const int m = m0 + 16 + r + 8 * ksel;
      const size_t o = (size_t)m * NHID + nl;
      float msb = sig1[m] * sigmoidf_(a1_1[r] + bi1);
      float ms  = sig2[m] * sigmoidf_(a2_1[r] + bi2);
      float t   = tanhf_(ayc1[r] + biy);
      float zn  = (1.0f - ms) * z[o] + ms * t;
      z_new[o]  = zn;
      zh[o]     = (_Float16)zn;
      iz_st[o]  = (_Float16)(iz_1[r] + biz);
      msb_st[o] = (_Float16)msb;
    }
  }
}

// ---------------- output GEMM: v = z_new@Wz.T, y_new fuse ---------------
__global__ __launch_bounds__(256) void lem_out_kernel(
    const _Float16* __restrict__ zh, const _Float16* __restrict__ Wz,
    const float* __restrict__ yold, const _Float16* __restrict__ iz_st,
    const _Float16* __restrict__ msb_st, float* __restrict__ y_new) {
  __shared__ __align__(16) _Float16 sB[2][16][KSTEP];  // 2 KB ping-pong
  v4u* sbq = (v4u*)&sB[0][0][0];  // 64 v4u chunks per buffer

  const int tid  = threadIdx.x;
  const int wave = tid >> 5;
  const int lane = tid & 31;
  const int ksel = lane >> 4;
  const int lrow = lane & 15;

  const int n0 = blockIdx.y * 16;
  const int m0 = blockIdx.x * 256 + wave * 32;

  const bool stager = tid < 64;
  const int srow = tid >> 2;
  const _Float16* p0 =
      Wz + (size_t)(n0 + (stager ? srow : 0)) * NHID + (tid & 3) * 8;

  const _Float16* zr0 = zh + (size_t)(m0 + lrow) * NHID + ksel * 8;
  const _Float16* zr1 = zr0 + 16 * NHID;

  if (stager) sbq[tid] = ldq(p0);
  __syncthreads();

  v16h az0 = make_frag(ldq(zr0), ldq(zr0 + 16));
  v16h az1 = make_frag(ldq(zr1), ldq(zr1 + 16));
  v8f v0 = {}; v8f v1 = {};

  int buf = 0;
  const int NIT = NHID / KSTEP;
  for (int it = 0; it < NIT; ++it) {
    const bool more = (it + 1) < NIT;
    v4u t0;
    v16h nz0, nz1;
    if (more) {
      p0 += KSTEP;
      if (stager) t0 = ldq(p0);
      zr0 += KSTEP; zr1 += KSTEP;
      nz0 = make_frag(ldq(zr0), ldq(zr0 + 16));
      nz1 = make_frag(ldq(zr1), ldq(zr1 + 16));
    }

    const int bb = buf * 64 + lrow * 4 + ksel;
    v16h b = make_frag(sbq[bb], sbq[bb + 2]);
    v0 = wmma_(az0, b, v0);
    v1 = wmma_(az1, b, v1);

    if (more) {
      if (stager) sbq[(buf ^ 1) * 64 + tid] = t0;
      az0 = nz0; az1 = nz1;
    }
    __syncthreads();
    buf ^= 1;
  }

  const int nl = n0 + lrow;
#pragma unroll
  for (int r = 0; r < 8; ++r) {
    {
      const int m = m0 + r + 8 * ksel;
      const size_t o = (size_t)m * NHID + nl;
      float msb = (float)msb_st[o];
      float t   = tanhf_(v0[r] + (float)iz_st[o]);
      y_new[o]  = (1.0f - msb) * yold[o] + msb * t;
    }
    {
      const int m = m0 + 16 + r + 8 * ksel;
      const size_t o = (size_t)m * NHID + nl;
      float msb = (float)msb_st[o];
      float t   = tanhf_(v1[r] + (float)iz_st[o]);
      y_new[o]  = (1.0f - msb) * yold[o] + msb * t;
    }
  }
}

// ------------------------------ launcher --------------------------------
extern "C" void kernel_launch(void* const* d_in, const int* in_sizes, int n_in,
                              void* d_out, int out_size, void* d_ws, size_t ws_size,
                              hipStream_t stream) {
  (void)in_sizes; (void)n_in; (void)out_size; (void)ws_size;

  const float* x    = (const float*)d_in[0];
  const float* y    = (const float*)d_in[1];
  const float* z    = (const float*)d_in[2];
  const float* dt   = (const float*)d_in[3];
  const float* Wih  = (const float*)d_in[4];
  const float* bih  = (const float*)d_in[5];
  const float* Whh  = (const float*)d_in[6];
  const float* bhh  = (const float*)d_in[7];
  const float* Wz   = (const float*)d_in[8];
  const float* bz   = (const float*)d_in[9];
  const float* Wdt  = (const float*)d_in[10];
  const float* bdt  = (const float*)d_in[11];

  float* y_new = (float*)d_out;
  float* z_new = y_new + (size_t)BATCH * NHID;

  char* ws = (char*)d_ws;
  size_t off = 0;
  _Float16* xh    = (_Float16*)(ws + off); off += (size_t)BATCH * NINP * 2;
  _Float16* yh    = (_Float16*)(ws + off); off += (size_t)BATCH * NHID * 2;
  _Float16* Wihh  = (_Float16*)(ws + off); off += (size_t)4 * NHID * NINP * 2;
  _Float16* Whhh  = (_Float16*)(ws + off); off += (size_t)3 * NHID * NHID * 2;
  _Float16* Wzh   = (_Float16*)(ws + off); off += (size_t)NHID * NHID * 2;
  _Float16* zh    = (_Float16*)(ws + off); off += (size_t)BATCH * NHID * 2;
  _Float16* izst  = (_Float16*)(ws + off); off += (size_t)BATCH * NHID * 2;
  _Float16* msbst = (_Float16*)(ws + off); off += (size_t)BATCH * NHID * 2;
  float*    sig1  = (float*)(ws + off);    off += (size_t)BATCH * 4;
  float*    sig2  = (float*)(ws + off);    off += (size_t)BATCH * 4;

  const int T = 256;
  auto g4 = [](size_t n4) { return (unsigned)((n4 + 255) / 256); };

  cvt_f32_f16_x4<<<g4((size_t)BATCH * NINP / 4), T, 0, stream>>>(x,   xh,   BATCH * NINP / 4);
  cvt_f32_f16_x4<<<g4((size_t)BATCH * NHID / 4), T, 0, stream>>>(y,   yh,   BATCH * NHID / 4);
  cvt_f32_f16_x4<<<g4((size_t)4 * NHID * NINP / 4), T, 0, stream>>>(Wih, Wihh, 4 * NHID * NINP / 4);
  cvt_f32_f16_x4<<<g4((size_t)3 * NHID * NHID / 4), T, 0, stream>>>(Whh, Whhh, 3 * NHID * NHID / 4);
  cvt_f32_f16_x4<<<g4((size_t)NHID * NHID / 4), T, 0, stream>>>(Wz,  Wzh,  NHID * NHID / 4);

  dt_gate_kernel<<<(BATCH + T - 1) / T, T, 0, stream>>>(dt, Wdt, bdt, sig1, sig2);

  dim3 grid(BATCH / 256, NHID / 16);
  lem_gates_kernel<<<grid, T, 0, stream>>>(xh, yh, Wihh, Whhh, bih, bhh, bz, z,
                                           sig1, sig2, z_new, zh, izst, msbst);
  lem_out_kernel<<<grid, T, 0, stream>>>(zh, Wzh, y, izst, msbst, y_new);
}
